// GNNImageClassificator_9543417331752
// MI455X (gfx1250) — matrix operand order repaired and checked by reading, not compiled, not run on hardware
//
#include <hip/hip_runtime.h>
#include <hip/hip_bf16.h>

#define Bg   64
#define Nn   1024
#define Ed   8192
#define EPE  (Ed + Nn)     /* 9216 edges incl. self loops */
#define Hh   152
#define Hp   160           /* padded hidden (10 tiles of 16) */
#define HpD  (Hp / 2)
#define Cc   3
#define KTt  5             /* k-tiles of 32 */
#define CTt  10            /* column tiles of 16 */
#define FCIN 611
#define LDSW 88            /* padded LDS row stride in dwords (16B aligned frags) */

typedef __attribute__((ext_vector_type(16))) __bf16 bf16x16;
typedef __attribute__((ext_vector_type(8)))  float  f32x8;
typedef __attribute__((ext_vector_type(4)))  unsigned u32x4;

union Frag { unsigned u[8]; u32x4 q[2]; bf16x16 v; };

__device__ __forceinline__ unsigned f2bf(float f) {
  unsigned x = __float_as_uint(f);
  unsigned r = (x + 0x7fffu + ((x >> 16) & 1u)) >> 16;   // round-to-nearest-even
  return r & 0xffffu;
}

__device__ __forceinline__ float atomicMaxF(float* addr, float val) {
  unsigned* ua = (unsigned*)addr;
  unsigned old = *ua;
  while (__uint_as_float(old) < val) {
    unsigned assumed = old;
    old = atomicCAS(ua, assumed, __float_as_uint(val));
    if (old == assumed) break;
  }
  return __uint_as_float(old);
}

// ---------------- layer 1 linear (K=3, VALU) ----------------
__global__ void k_lin1(const float* __restrict__ x, const float* __restrict__ W1,
                       float* __restrict__ hlin) {
  size_t t = (size_t)blockIdx.x * blockDim.x + threadIdx.x;
  size_t total = (size_t)Bg * Nn * Hp;
  if (t >= total) return;
  int j = (int)(t % Hp);
  size_t bn = t / Hp;
  float v = 0.f;
  if (j < Hh) {
    const float* xp = x + bn * Cc;
    v = xp[0] * W1[j] + xp[1] * W1[Hh + j] + xp[2] * W1[2 * Hh + j];
  }
  hlin[t] = v;
}

// ---------------- f32 -> bf16 convert (zero-pad cols >= 152) ----------------
__global__ void k_convert(const float* __restrict__ hin, unsigned* __restrict__ hbf) {
  size_t t = (size_t)blockIdx.x * blockDim.x + threadIdx.x;
  size_t total = (size_t)Bg * Nn * HpD;
  if (t >= total) return;
  int jd = (int)(t % HpD);
  size_t bn = t / HpD;
  int j0 = jd * 2;
  const float* r = hin + bn * Hp;
  unsigned lo = (j0     < Hh) ? f2bf(r[j0])     : 0u;
  unsigned hi = (j0 + 1 < Hh) ? f2bf(r[j0 + 1]) : 0u;
  hbf[t] = lo | (hi << 16);
}

// ---------------- pack W[152x152] into WMMA B-fragment layout ----------------
// wpack index: ((ct*KTt + kt)*32 + lane)*8 + v ; lane: n=ct*16+(lane&15), half=lane>>4
// VGPR v holds K = kt*32 + half*16 + 2v (+1 in high half)
__global__ void k_pack_w(const float* __restrict__ W, unsigned* __restrict__ wpack) {
  int t = blockIdx.x * blockDim.x + threadIdx.x;
  if (t >= CTt * KTt * 32 * 8) return;
  int v    = t & 7;
  int lane = (t >> 3) & 31;
  int rest = t >> 8;
  int kt   = rest % KTt;
  int ct   = rest / KTt;
  int n  = ct * 16 + (lane & 15);
  int k0 = kt * 32 + (lane >> 4) * 16 + 2 * v;
  unsigned lo = (k0     < Hh && n < Hh) ? f2bf(W[(size_t)k0 * Hh + n])       : 0u;
  unsigned hi = (k0 + 1 < Hh && n < Hh) ? f2bf(W[(size_t)(k0 + 1) * Hh + n]) : 0u;
  wpack[t] = lo | (hi << 16);
}

// ---------------- WMMA GEMM: hlin[b] = hbf[b] (1024xHp) * W (HpxHp) ----------------
// block = (graph, row tile); 10 waves, wave w owns column tile w.
// The shared 16x160 bf16 A tile is staged once per block into LDS with the
// CDNA5 async-to-LDS path (ASYNCcnt), then waves read fragments via ds_load.
__global__ void __launch_bounds__(320, 1)
k_gemm(const unsigned* __restrict__ hbf, const unsigned* __restrict__ wpack,
       float* __restrict__ hlin) {
  __shared__ __align__(16) unsigned atile[16 * LDSW];

  // Escape atile's address into opaque asm with a memory clobber: the compiler
  // must now assume later volatile asm (the async load below) can write it, so
  // the array is kept allocated and the ds_loads are not folded away.
  asm volatile("" :: "v"(&atile[0]) : "memory");

  int mt   = blockIdx.x & 63;
  int b    = blockIdx.x >> 6;
  int t    = threadIdx.x;
  int w    = t >> 5;
  int lane = t & 31;

  // ---- async stage: 16 rows x 80 dwords, exactly one b128 per thread ----
  {
    int row   = t / 20;            // 0..15
    int chunk = t - row * 20;      // 0..19 (4-dword chunks)
    unsigned lds_off = (unsigned)((row * LDSW + chunk * 4) * 4);   // atile at LDS offset 0
    const unsigned* g = hbf + ((size_t)b * Nn + (size_t)mt * 16 + row) * HpD + chunk * 4;
    unsigned long long ga = (unsigned long long)g;
    asm volatile("global_load_async_to_lds_b128 %0, %1, off"
                 :: "v"(lds_off), "v"(ga)
                 : "memory");
  }
#if __has_builtin(__builtin_amdgcn_s_wait_asynccnt)
  __builtin_amdgcn_s_wait_asynccnt(0);
#else
  asm volatile("s_wait_asynccnt 0x0" ::: "memory");
#endif
  __syncthreads();

  int m    = lane & 15;
  int half = lane >> 4;

  f32x8 acc = {};
#pragma unroll
  for (int kt = 0; kt < KTt; ++kt) {
    Frag af, bw;
    // A fragment from LDS: v=0..7 -> dwords kd+half*8+0..7
    const u32x4* ap = (const u32x4*)&atile[m * LDSW + kt * 16 + half * 8];
    af.q[0] = ap[0];
    af.q[1] = ap[1];
    // B fragment: packed contiguous in global (tiny, L2-resident)
    const u32x4* bp = (const u32x4*)(wpack + (((size_t)w * KTt + kt) * 32 + lane) * 8);
    bw.q[0] = bp[0];
    bw.q[1] = bp[1];
    acc = __builtin_amdgcn_wmma_f32_16x16x32_bf16(false, af.v, false, bw.v,
                                                  (short)0, acc, false, false);
  }
  // C layout: lane n = w*16 + (lane&15); VGPR r -> row m = half*8 + r
  int n = w * 16 + (lane & 15);
  float* crow = hlin + ((size_t)b * Nn + (size_t)mt * 16) * Hp + n;
#pragma unroll
  for (int r = 0; r < 8; ++r) {
    int mm = half * 8 + r;
    crow[(size_t)mm * Hp] = acc[r];
  }
}

// ---------------- per-node attention scores (wave per node) ----------------
__global__ void k_scores(const float* __restrict__ hlin, const float* __restrict__ a_s,
                         const float* __restrict__ a_d, float* __restrict__ ss,
                         float* __restrict__ sd) {
  int wid  = blockIdx.x * (blockDim.x >> 5) + (threadIdx.x >> 5);
  int lane = threadIdx.x & 31;
  if (wid >= Bg * Nn) return;
  const float* h = hlin + (size_t)wid * Hp;
  float s1 = 0.f, s2 = 0.f;
  for (int j = lane; j < Hh; j += 32) {
    float v = h[j];
    s1 += v * a_s[j];
    s2 += v * a_d[j];
  }
  for (int off = 16; off; off >>= 1) {
    s1 += __shfl_down(s1, off);
    s2 += __shfl_down(s2, off);
  }
  if (lane == 0) { ss[wid] = s1; sd[wid] = s2; }
}

// ---------------- init: hout = bias (cols<152) / 0, mmax=-inf, denom=0 ----------------
__global__ void k_init(float* __restrict__ hout, const float* __restrict__ bias,
                       float* __restrict__ mmax, float* __restrict__ denom) {
  size_t t = (size_t)blockIdx.x * blockDim.x + threadIdx.x;
  size_t total = (size_t)Bg * Nn * Hp;
  if (t >= total) return;
  int j = (int)(t % Hp);
  size_t bn = t / Hp;
  hout[t] = (j < Hh) ? bias[j] : 0.f;
  if (j == 0) { mmax[bn] = -3.0e38f; denom[bn] = 0.f; }
}

__device__ __forceinline__ void edge_sd(const int* __restrict__ eidx, int b, int e,
                                        int& s, int& d) {
  if (e < Ed) {
    const int* eg = eidx + (size_t)b * 2 * Ed;
    s = eg[e];
    d = eg[Ed + e];
  } else {
    s = d = e - Ed;    // self loop
  }
}

__global__ void k_edge_max(const int* __restrict__ eidx, const float* __restrict__ ss,
                           const float* __restrict__ sd, float* __restrict__ mmax) {
  int t = blockIdx.x * blockDim.x + threadIdx.x;
  if (t >= Bg * EPE) return;
  int b = t / EPE, e = t - b * EPE;
  int s, d; edge_sd(eidx, b, e, s, d);
  float ev = ss[b * Nn + s] + sd[b * Nn + d];
  ev = ev > 0.f ? ev : 0.2f * ev;                 // leaky_relu(0.2)
  atomicMaxF(&mmax[b * Nn + d], ev);
}

__global__ void k_edge_exp(const int* __restrict__ eidx, const float* __restrict__ ss,
                           const float* __restrict__ sd, const float* __restrict__ mmax,
                           float* __restrict__ denom, float* __restrict__ exn) {
  int t = blockIdx.x * blockDim.x + threadIdx.x;
  if (t >= Bg * EPE) return;
  int b = t / EPE, e = t - b * EPE;
  int s, d; edge_sd(eidx, b, e, s, d);
  float ev = ss[b * Nn + s] + sd[b * Nn + d];
  ev = ev > 0.f ? ev : 0.2f * ev;
  float ex = __expf(ev - mmax[b * Nn + d]);
  exn[t] = ex;
  atomicAdd(&denom[b * Nn + d], ex);
}

// ---------------- weighted scatter: hout[d] += alpha * hlin[s] (wave per edge) ----------------
__global__ void k_edge_aggr(const int* __restrict__ eidx, const float* __restrict__ hlin,
                            const float* __restrict__ exn, const float* __restrict__ denom,
                            float* __restrict__ hout) {
  int wid  = blockIdx.x * (blockDim.x >> 5) + (threadIdx.x >> 5);
  int lane = threadIdx.x & 31;
  if (wid >= Bg * EPE) return;
  int b = wid / EPE, e = wid - b * EPE;
  int s, d; edge_sd(eidx, b, e, s, d);
  float alpha = exn[wid] / denom[b * Nn + d];
  const float* hs = hlin + ((size_t)b * Nn + s) * Hp;
  float* ho = hout + ((size_t)b * Nn + d) * Hp;
  for (int j = lane; j < Hh; j += 32) atomicAdd(&ho[j], alpha * hs[j]);
}

// ---------------- column mean pool over nodes (wave per (b, feature)) ----------------
__global__ void k_pool(const float* __restrict__ src, int stride, int nfeat,
                       float* __restrict__ pooled, int offset) {
  int wid  = blockIdx.x * (blockDim.x >> 5) + (threadIdx.x >> 5);
  int lane = threadIdx.x & 31;
  if (wid >= Bg * nfeat) return;
  int b = wid / nfeat, j = wid - b * nfeat;
  const float* p = src + (size_t)b * Nn * stride + j;
  float sum = 0.f;
  for (int n = lane; n < Nn; n += 32) sum += p[(size_t)n * stride];
  for (int off = 16; off; off >>= 1) sum += __shfl_down(sum, off);
  if (lane == 0) pooled[b * FCIN + offset + j] = sum * (1.0f / Nn);
}

// ---------------- tiny FC layers ----------------
__global__ void k_fc(const float* __restrict__ in, const float* __restrict__ W,
                     const float* __restrict__ bias, float* __restrict__ out,
                     int K, int Nout) {
  int t = blockIdx.x * blockDim.x + threadIdx.x;
  if (t >= Bg * Nout) return;
  int b = t / Nout, j = t - b * Nout;
  const float* ip = in + (size_t)b * K;
  float s = bias[j];
  for (int k = 0; k < K; ++k) s += ip[k] * W[(size_t)k * Nout + j];
  out[t] = s;
}

static inline int cdiv(long long a, int b) { return (int)((a + b - 1) / b); }

extern "C" void kernel_launch(void* const* d_in, const int* in_sizes, int n_in,
                              void* d_out, int out_size, void* d_ws, size_t ws_size,
                              hipStream_t stream) {
  const float* x     = (const float*)d_in[0];
  const int*   eidx  = (const int*)d_in[1];
  const float* W1    = (const float*)d_in[2];
  const float* as1   = (const float*)d_in[3];
  const float* ad1   = (const float*)d_in[4];
  const float* b1    = (const float*)d_in[5];
  const float* W234  = (const float*)d_in[6];
  const float* as234 = (const float*)d_in[7];
  const float* ad234 = (const float*)d_in[8];
  const float* b234  = (const float*)d_in[9];
  const float* fc1W  = (const float*)d_in[10];
  const float* fc1b  = (const float*)d_in[11];
  const float* fc2W  = (const float*)d_in[12];
  const float* fc2b  = (const float*)d_in[13];
  const float* fc3W  = (const float*)d_in[14];
  const float* fc3b  = (const float*)d_in[15];
  float* out = (float*)d_out;

  // workspace carve-up
  char* ws = (char*)d_ws;
  size_t off = 0;
  auto alloc = [&](size_t bytes) -> void* {
    void* p = ws + off;
    off = (off + bytes + 255) & ~(size_t)255;
    return p;
  };
  const size_t BN = (size_t)Bg * Nn;
  float*    HL     = (float*)alloc(BN * Hp * sizeof(float));      // GEMM / lin output
  float*    bufA   = (float*)alloc(BN * Hp * sizeof(float));      // node features ping
  float*    bufB   = (float*)alloc(BN * Hp * sizeof(float));      // node features pong
  unsigned* hbf    = (unsigned*)alloc(BN * HpD * sizeof(unsigned));
  unsigned* wpack  = (unsigned*)alloc((size_t)CTt * KTt * 32 * 8 * sizeof(unsigned));
  float*    ss     = (float*)alloc(BN * sizeof(float));
  float*    sd     = (float*)alloc(BN * sizeof(float));
  float*    mmax   = (float*)alloc(BN * sizeof(float));
  float*    denom  = (float*)alloc(BN * sizeof(float));
  float*    exn    = (float*)alloc((size_t)Bg * EPE * sizeof(float));
  float*    pooled = (float*)alloc((size_t)Bg * FCIN * sizeof(float));
  float*    m1     = (float*)alloc((size_t)Bg * 256 * sizeof(float));
  float*    m2     = (float*)alloc((size_t)Bg * 128 * sizeof(float));

  const int TB = 256;
  const long long totHp  = (long long)Bg * Nn * Hp;
  const long long totCvt = (long long)Bg * Nn * HpD;
  const long long totE   = (long long)Bg * EPE;

  auto attention = [&](const float* a_s, const float* a_d, const float* bias, float* hout) {
    k_scores  <<<cdiv((long long)Bg * Nn * 32, TB), TB, 0, stream>>>(HL, a_s, a_d, ss, sd);
    k_init    <<<cdiv(totHp, TB), TB, 0, stream>>>(hout, bias, mmax, denom);
    k_edge_max<<<cdiv(totE, TB),  TB, 0, stream>>>(eidx, ss, sd, mmax);
    k_edge_exp<<<cdiv(totE, TB),  TB, 0, stream>>>(eidx, ss, sd, mmax, denom, exn);
    k_edge_aggr<<<cdiv(totE * 32, TB), TB, 0, stream>>>(eidx, HL, exn, denom, hout);
  };

  // ---- layer 1 (K=3 linear, then attention) ----
  k_lin1<<<cdiv(totHp, TB), TB, 0, stream>>>(x, W1, HL);
  attention(as1, ad1, b1, bufA);
  k_pool<<<cdiv((long long)Bg * Hh * 32, TB), TB, 0, stream>>>(bufA, Hp, Hh, pooled, 3);

  // ---- layers 2..4 (WMMA GEMM + attention) ----
  float* hin  = bufA;
  float* hdst = bufB;
  for (int l = 0; l < 3; ++l) {
    k_convert<<<cdiv(totCvt, TB), TB, 0, stream>>>(hin, hbf);
    k_pack_w <<<cdiv((long long)CTt * KTt * 32 * 8, TB), TB, 0, stream>>>(
        W234 + (size_t)l * Hh * Hh, wpack);
    k_gemm   <<<Bg * 64, 320, 0, stream>>>(hbf, wpack, HL);
    attention(as234 + l * Hh, ad234 + l * Hh, b234 + l * Hh, hdst);
    k_pool<<<cdiv((long long)Bg * Hh * 32, TB), TB, 0, stream>>>(
        hdst, Hp, Hh, pooled, 3 + (l + 1) * Hh);
    float* tmp = hdst; hdst = hin; hin = tmp;
  }

  // ---- input-feature mean pool ----
  k_pool<<<cdiv((long long)Bg * Cc * 32, TB), TB, 0, stream>>>(x, Cc, Cc, pooled, 0);

  // ---- MLP head ----
  k_fc<<<cdiv((long long)Bg * 256, TB), TB, 0, stream>>>(pooled, fc1W, fc1b, m1, FCIN, 256);
  k_fc<<<cdiv((long long)Bg * 128, TB), TB, 0, stream>>>(m1, fc2W, fc2b, m2, 256, 128);
  k_fc<<<cdiv((long long)Bg * 10,  TB), TB, 0, stream>>>(m2, fc3W, fc3b, out, 128, 10);
}
